// EncoderLayer_64939905516364
// MI455X (gfx1250) — matrix operand order repaired and checked by reading, not compiled, not run on hardware
//
#include <hip/hip_runtime.h>

// ---------------------------------------------------------------------------
// Transformer encoder layer for MI455X (gfx1250, wave32, WMMA).
// All GEMMs / attention matmuls use v_wmma_f32_16x16x32_bf16.
// Tile staging uses GLOBAL_LOAD_ASYNC_TO_LDS (ASYNCcnt) where the toolchain
// exposes the builtins, with a VGPR-path fallback; GEMM is LDS double-buffered
// so async copies overlap WMMA compute.
// ---------------------------------------------------------------------------

#define D_MODEL 1024
#define SEQ     2048
#define BATCH   2
#define NHEAD   16
#define HEAD_DIM 64
#define D_FF    2048
#define EPSF    1e-6f

typedef __bf16 bf16;
typedef __attribute__((ext_vector_type(8)))  bf16  bf16x8;
typedef __attribute__((ext_vector_type(16))) bf16  bf16x16;
typedef __attribute__((ext_vector_type(8)))  float f32x8;
typedef __attribute__((ext_vector_type(4)))  int   v4i;

typedef __attribute__((address_space(1))) v4i gv4i;   // global int4
typedef __attribute__((address_space(3))) v4i sv4i;   // LDS int4

union FragAB { bf16x16 v; bf16x8 h[2]; };

#if defined(__has_builtin)
#if __has_builtin(__builtin_amdgcn_global_load_async_to_lds_b128)
#define HAVE_ASYNC_LDS 1
#endif
#endif
#ifndef HAVE_ASYNC_LDS
#define HAVE_ASYNC_LDS 0
#endif

// 16-byte global -> LDS copy. Async path: DMA into LDS, tracked by ASYNCcnt,
// no VGPR round trip. Fallback: b128 load + ds_store_b128.
__device__ __forceinline__ void copy16_g2lds(void* lds_dst, const void* gsrc) {
#if HAVE_ASYNC_LDS
  __builtin_amdgcn_global_load_async_to_lds_b128(
      (gv4i*)(uintptr_t)gsrc,
      (sv4i*)(uint32_t)(uintptr_t)lds_dst,
      0, 0);
#else
  *(uint4*)lds_dst = *(const uint4*)gsrc;
#endif
}
__device__ __forceinline__ void async_wait() {
#if HAVE_ASYNC_LDS
  asm volatile("s_wait_asynccnt 0" ::: "memory");
#endif
}

__device__ __forceinline__ f32x8 zero8() {
  f32x8 z;
#pragma unroll
  for (int i = 0; i < 8; ++i) z[i] = 0.f;
  return z;
}

__device__ __forceinline__ f32x8 wmma_bf16(const FragAB& a, const FragAB& b, f32x8 c) {
  // D = A(16x32 bf16) * B(32x16 bf16) + C(16x16 f32)
  return __builtin_amdgcn_wmma_f32_16x16x32_bf16(false, a.v, false, b.v,
                                                 (short)0, c, false, false);
}

// ---------------------------------------------------------------------------
// f32 -> bf16 conversion
// ---------------------------------------------------------------------------
__global__ void cvt_f32_to_bf16(const float* __restrict__ in,
                                bf16* __restrict__ out, int n) {
  int i = blockIdx.x * blockDim.x + threadIdx.x;
  if (i < n) out[i] = (bf16)in[i];
}

// ---------------------------------------------------------------------------
// GEMM: C[M,N] = A[M,K](bf16) * W[N,K]^T(bf16) + bias[N], optional ReLU,
// optional scale, output f32 or bf16.  Block tile 128x128, 8 waves, each wave
// 4x2 WMMA 16x16 tiles.  K-step 32, LDS double-buffered: async loads for tile
// kt+1 are issued before computing tile kt (one barrier per step).
// ---------------------------------------------------------------------------
template<bool RELU, bool OUT_BF16>
__global__ __launch_bounds__(256) void gemm_xwT(
    const bf16* __restrict__ A, const bf16* __restrict__ W,
    const float* __restrict__ bias, float scale, void* __restrict__ Cout,
    int M, int N, int K)
{
  (void)M;
  __shared__ alignas(16) bf16 sA [2][128][40];    // row-major A tile (pad 8)
  __shared__ alignas(16) bf16 sBt[2][32][136];    // transposed W tile: [k][n]

  const int t    = threadIdx.x;
  const int wave = t >> 5, lane = t & 31;
  const int hl   = lane >> 4, ln = lane & 15;
  const int wm   = (wave >> 2) * 64, wn = (wave & 3) * 32;
  const int mBase = blockIdx.y * 128, nBase = blockIdx.x * 128;

  const int lrow = t >> 1;         // 0..127
  const int lc0  = (t & 1) * 16;   // 0 or 16

  f32x8 acc[4][2];
#pragma unroll
  for (int mi = 0; mi < 4; ++mi)
#pragma unroll
    for (int ni = 0; ni < 2; ++ni) acc[mi][ni] = zero8();

  auto load_tile = [&](int k0, int buf) {
    // A tile: 32B per thread, contiguous -> async DMA to LDS
    const bf16* ga = A + (size_t)(mBase + lrow) * K + k0 + lc0;
    copy16_g2lds(&sA[buf][lrow][lc0],     ga);
    copy16_g2lds(&sA[buf][lrow][lc0 + 8], ga + 8);
    // W tile, stored transposed as [k][n]; B-fragments gathered one bf16 per k row
    union { uint4 u[2]; bf16 b[16]; } tw;
    const uint4* g = (const uint4*)(W + (size_t)(nBase + lrow) * K + k0 + lc0);
    tw.u[0] = g[0]; tw.u[1] = g[1];
#pragma unroll
    for (int j = 0; j < 16; ++j) sBt[buf][lc0 + j][lrow] = tw.b[j];
  };

  const int nk = K >> 5;
  load_tile(0, 0);

  for (int kt = 0; kt < nk; ++kt) {
    const int cur = kt & 1;
    async_wait();          // tile kt's async copies done (DS waits by barrier)
    __syncthreads();
    if (kt + 1 < nk) load_tile((kt + 1) << 5, cur ^ 1);  // overlap with WMMA

    FragAB bfr[2];
#pragma unroll
    for (int ni = 0; ni < 2; ++ni) {   // B: lane&15 = N col, K in table order by half
#pragma unroll
      for (int s = 0; s < 8; ++s) {
        bfr[ni].v[s]     = sBt[cur][hl * 8 + s][wn + ni * 16 + ln];
        bfr[ni].v[8 + s] = sBt[cur][16 + hl * 8 + s][wn + ni * 16 + ln];
      }
    }
#pragma unroll
    for (int mi = 0; mi < 4; ++mi) {   // A: lane = M row, K chunks by half
      FragAB afr;
      const int r = wm + mi * 16 + ln;
      afr.h[0] = *(const bf16x8*)&sA[cur][r][hl * 8];
      afr.h[1] = *(const bf16x8*)&sA[cur][r][16 + hl * 8];
#pragma unroll
      for (int ni = 0; ni < 2; ++ni)
        acc[mi][ni] = wmma_bf16(afr, bfr[ni], acc[mi][ni]);
    }
  }

  // Epilogue: C layout — VGPR r -> row r + 8*half, col = lane&15
#pragma unroll
  for (int mi = 0; mi < 4; ++mi)
#pragma unroll
    for (int ni = 0; ni < 2; ++ni) {
      const int col = nBase + wn + ni * 16 + ln;
      const float bv = bias[col];
#pragma unroll
      for (int r = 0; r < 8; ++r) {
        const int row = mBase + wm + mi * 16 + r + hl * 8;
        float v = acc[mi][ni][r] + bv;
        if (RELU) v = fmaxf(v, 0.f);
        v *= scale;
        if (OUT_BF16) ((bf16*)Cout)[(size_t)row * N + col] = (bf16)v;
        else          ((float*)Cout)[(size_t)row * N + col] = v;
      }
    }
}

// ---------------------------------------------------------------------------
// Flash-style attention.  Q already scaled by 1/sqrt(HEAD_DIM).
// grid = (SEQ/64, BATCH*NHEAD), block = 128 (4 waves, 16 q-rows each).
// ---------------------------------------------------------------------------
__global__ __launch_bounds__(128) void attention_kernel(
    const bf16* __restrict__ Q, const bf16* __restrict__ Kc,
    const bf16* __restrict__ V, const int* __restrict__ mask,
    bf16* __restrict__ Ctx)
{
  __shared__ alignas(16) bf16 sQ [64][72];
  __shared__ alignas(16) bf16 sKt[64][72];   // [d][key]
  __shared__ alignas(16) bf16 sV [64][72];   // [key][d]
  __shared__ alignas(16) bf16 sP [64][72];   // probs, wave-private rows

  const int t    = threadIdx.x;
  const int wave = t >> 5, lane = t & 31;
  const int hl   = lane >> 4, ln = lane & 15;
  const int wq   = wave * 16;

  const int qBase = blockIdx.x * 64;
  const int bh = blockIdx.y;
  const int b = bh >> 4, h = bh & 15;
  const size_t rowOff   = (size_t)b * SEQ;
  const size_t maskBase = (size_t)b * SEQ * SEQ;

  const int lrow = t >> 1;        // 0..63
  const int lc0  = (t & 1) * 32;  // 0 or 32

  {   // load Q tile (64x64 bf16) via async DMA
    const bf16* g = Q + (rowOff + qBase + lrow) * D_MODEL + h * HEAD_DIM + lc0;
#pragma unroll
    for (int c = 0; c < 4; ++c)
      copy16_g2lds(&sQ[lrow][lc0 + c * 8], g + c * 8);
  }
  async_wait();
  __syncthreads();

  FragAB aq[2];
#pragma unroll
  for (int ks = 0; ks < 2; ++ks) {
    aq[ks].h[0] = *(const bf16x8*)&sQ[wq + ln][ks * 32 + hl * 8];
    aq[ks].h[1] = *(const bf16x8*)&sQ[wq + ln][ks * 32 + 16 + hl * 8];
  }

  f32x8 o[4];
  float rowM[8], rowS[8];
#pragma unroll
  for (int di = 0; di < 4; ++di) o[di] = zero8();
#pragma unroll
  for (int r = 0; r < 8; ++r) { rowM[r] = -1e30f; rowS[r] = 0.f; }

  for (int kt0 = 0; kt0 < SEQ; kt0 += 64) {
    __syncthreads();
    {   // K tile transposed into sKt (manual), V tile async into sV
      union { uint4 u[4]; bf16 e[32]; } tw;
      const uint4* gk = (const uint4*)(Kc + (rowOff + kt0 + lrow) * D_MODEL
                                          + h * HEAD_DIM + lc0);
      tw.u[0] = gk[0]; tw.u[1] = gk[1]; tw.u[2] = gk[2]; tw.u[3] = gk[3];
#pragma unroll
      for (int j = 0; j < 32; ++j) sKt[lc0 + j][lrow] = tw.e[j];
      const bf16* gv = V + (rowOff + kt0 + lrow) * D_MODEL + h * HEAD_DIM + lc0;
#pragma unroll
      for (int c = 0; c < 4; ++c)
        copy16_g2lds(&sV[lrow][lc0 + c * 8], gv + c * 8);
    }
    async_wait();
    __syncthreads();

    // S = Q * K^T  (4 key sub-tiles of 16, K-dim 64 = 2 WMMA steps)
    f32x8 sc[4];
#pragma unroll
    for (int ni = 0; ni < 4; ++ni) {
      sc[ni] = zero8();
#pragma unroll
      for (int ks = 0; ks < 2; ++ks) {
        FragAB bk;
#pragma unroll
        for (int s = 0; s < 8; ++s) {
          bk.v[s]     = sKt[ks * 32 + hl * 8 + s][ni * 16 + ln];
          bk.v[8 + s] = sKt[ks * 32 + 16 + hl * 8 + s][ni * 16 + ln];
        }
        sc[ni] = wmma_bf16(aq[ks], bk, sc[ni]);
      }
    }
    // mask
#pragma unroll
    for (int ni = 0; ni < 4; ++ni) {
      const int col = kt0 + ni * 16 + ln;
#pragma unroll
      for (int r = 0; r < 8; ++r) {
        const int rowq = qBase + wq + r + hl * 8;
        if (mask[maskBase + (size_t)rowq * SEQ + col] == 0) sc[ni][r] = -1e9f;
      }
    }
    // online softmax: reductions over the 16-lane halves (wave32-safe)
#pragma unroll
    for (int r = 0; r < 8; ++r) {
      float mx = fmaxf(fmaxf(sc[0][r], sc[1][r]), fmaxf(sc[2][r], sc[3][r]));
#pragma unroll
      for (int d = 1; d < 16; d <<= 1) mx = fmaxf(mx, __shfl_xor(mx, d, 32));
      const float mnew = fmaxf(rowM[r], mx);
      const float resc = __expf(rowM[r] - mnew);
      rowM[r] = mnew;
      float psum = 0.f;
      const int prow = wq + r + hl * 8;
#pragma unroll
      for (int ni = 0; ni < 4; ++ni) {
        const float p = __expf(sc[ni][r] - mnew);
        psum += p;
        sP[prow][ni * 16 + ln] = (bf16)p;
      }
#pragma unroll
      for (int d = 1; d < 16; d <<= 1) psum += __shfl_xor(psum, d, 32);
      rowS[r] = rowS[r] * resc + psum;
#pragma unroll
      for (int di = 0; di < 4; ++di) o[di][r] *= resc;
    }
    // same-wave DS store->load ordering: DS is in-order per wave; wait for
    // returned data before the A-fragment reads of sP.
    asm volatile("s_wait_dscnt 0" ::: "memory");

    // O += P * V
#pragma unroll
    for (int ks = 0; ks < 2; ++ks) {
      FragAB ap;
      ap.h[0] = *(const bf16x8*)&sP[wq + ln][ks * 32 + hl * 8];
      ap.h[1] = *(const bf16x8*)&sP[wq + ln][ks * 32 + 16 + hl * 8];
#pragma unroll
      for (int di = 0; di < 4; ++di) {
        FragAB bv;
#pragma unroll
        for (int s = 0; s < 8; ++s) {
          bv.v[s]     = sV[ks * 32 + hl * 8 + s][di * 16 + ln];
          bv.v[8 + s] = sV[ks * 32 + 16 + hl * 8 + s][di * 16 + ln];
        }
        o[di] = wmma_bf16(ap, bv, o[di]);
      }
    }
  }

#pragma unroll
  for (int di = 0; di < 4; ++di)
#pragma unroll
    for (int r = 0; r < 8; ++r) {
      const int rowq = qBase + wq + r + hl * 8;
      Ctx[(rowOff + rowq) * D_MODEL + h * HEAD_DIM + di * 16 + ln] =
          (bf16)(o[di][r] / rowS[r]);
    }
}

// ---------------------------------------------------------------------------
// y = alpha * (a + b - mean) / (std(ddof=1) + eps) + beta     (one row/block)
// ---------------------------------------------------------------------------
template<bool WRITE_BF16>
__global__ __launch_bounds__(256) void add_norm(
    const float* __restrict__ A, const float* __restrict__ Badd,
    const float* __restrict__ alpha, const float* __restrict__ beta,
    float* __restrict__ Yf, bf16* __restrict__ Yb)
{
  __shared__ float rs[256], rq[256];
  const int row = blockIdx.x, t = threadIdx.x;
  const size_t base = (size_t)row * D_MODEL;
  float4 xa = *(const float4*)(A + base + t * 4);
  float4 xb = *(const float4*)(Badd + base + t * 4);
  float x[4] = {xa.x + xb.x, xa.y + xb.y, xa.z + xb.z, xa.w + xb.w};
  float s = x[0] + x[1] + x[2] + x[3];
  float q = x[0]*x[0] + x[1]*x[1] + x[2]*x[2] + x[3]*x[3];
  rs[t] = s; rq[t] = q;
  __syncthreads();
  for (int off = 128; off > 0; off >>= 1) {
    if (t < off) { rs[t] += rs[t + off]; rq[t] += rq[t + off]; }
    __syncthreads();
  }
  const float mean = rs[0] * (1.f / D_MODEL);
  const float var  = (rq[0] - (float)D_MODEL * mean * mean) * (1.f / (D_MODEL - 1));
  const float inv  = 1.f / (sqrtf(fmaxf(var, 0.f)) + EPSF);
#pragma unroll
  for (int i = 0; i < 4; ++i) {
    const int c = t * 4 + i;
    const float y = alpha[c] * (x[i] - mean) * inv + beta[c];
    Yf[base + c] = y;
    if (WRITE_BF16) Yb[base + c] = (bf16)y;
  }
}

// ---------------------------------------------------------------------------
extern "C" void kernel_launch(void* const* d_in, const int* in_sizes, int n_in,
                              void* d_out, int out_size, void* d_ws, size_t ws_size,
                              hipStream_t stream) {
  (void)in_sizes; (void)n_in; (void)out_size; (void)ws_size;
  const float* src = (const float*)d_in[0];
  const int*   msk = (const int*)d_in[1];
  const float* Wq = (const float*)d_in[2];  const float* bq = (const float*)d_in[3];
  const float* Wk = (const float*)d_in[4];  const float* bk = (const float*)d_in[5];
  const float* Wv = (const float*)d_in[6];  const float* bv = (const float*)d_in[7];
  const float* Wo = (const float*)d_in[8];  const float* bo = (const float*)d_in[9];
  const float* W1 = (const float*)d_in[10]; const float* b1 = (const float*)d_in[11];
  const float* W2 = (const float*)d_in[12]; const float* b2 = (const float*)d_in[13];
  const float* alpha1 = (const float*)d_in[14]; const float* beta1 = (const float*)d_in[15];
  const float* alpha2 = (const float*)d_in[16]; const float* beta2 = (const float*)d_in[17];

  const int ROWS = BATCH * SEQ;  // 4096
  char* ws = (char*)d_ws;
  size_t off = 0;
  auto carve = [&](size_t bytes) -> char* {
    char* p = ws + off; off += (bytes + 255) & ~(size_t)255; return p;
  };
  bf16*  Xb   = (bf16*) carve((size_t)ROWS * D_MODEL * 2);
  bf16*  Wqb  = (bf16*) carve((size_t)D_MODEL * D_MODEL * 2);
  bf16*  Wkb  = (bf16*) carve((size_t)D_MODEL * D_MODEL * 2);
  bf16*  Wvb  = (bf16*) carve((size_t)D_MODEL * D_MODEL * 2);
  bf16*  Wob  = (bf16*) carve((size_t)D_MODEL * D_MODEL * 2);
  bf16*  W1b  = (bf16*) carve((size_t)D_FF * D_MODEL * 2);
  bf16*  W2b  = (bf16*) carve((size_t)D_MODEL * D_FF * 2);
  bf16*  Qb   = (bf16*) carve((size_t)ROWS * D_MODEL * 2);
  bf16*  Kb   = (bf16*) carve((size_t)ROWS * D_MODEL * 2);
  bf16*  Vb   = (bf16*) carve((size_t)ROWS * D_MODEL * 2);
  bf16*  Ctxb = (bf16*) carve((size_t)ROWS * D_MODEL * 2);
  float* attn = (float*)carve((size_t)ROWS * D_MODEL * 4);
  float* x1   = (float*)carve((size_t)ROWS * D_MODEL * 4);
  bf16*  x1b  = (bf16*) carve((size_t)ROWS * D_MODEL * 2);
  bf16*  Hb   = (bf16*) carve((size_t)ROWS * D_FF * 2);
  float* ff   = (float*)carve((size_t)ROWS * D_MODEL * 4);

  auto cvt = [&](const float* in, bf16* out, int n) {
    cvt_f32_to_bf16<<<n / 256, 256, 0, stream>>>(in, out, n);
  };
  cvt(src, Xb, ROWS * D_MODEL);
  cvt(Wq, Wqb, D_MODEL * D_MODEL);
  cvt(Wk, Wkb, D_MODEL * D_MODEL);
  cvt(Wv, Wvb, D_MODEL * D_MODEL);
  cvt(Wo, Wob, D_MODEL * D_MODEL);
  cvt(W1, W1b, D_FF * D_MODEL);
  cvt(W2, W2b, D_MODEL * D_FF);

  const dim3 blk(256);
  const dim3 gP(D_MODEL / 128, ROWS / 128);   // projections: (8, 32)
  const float qscale = 0.125f;                // 1/sqrt(HEAD_DIM)
  gemm_xwT<false, true ><<<gP, blk, 0, stream>>>(Xb, Wqb, bq, qscale, Qb, ROWS, D_MODEL, D_MODEL);
  gemm_xwT<false, true ><<<gP, blk, 0, stream>>>(Xb, Wkb, bk, 1.f,    Kb, ROWS, D_MODEL, D_MODEL);
  gemm_xwT<false, true ><<<gP, blk, 0, stream>>>(Xb, Wvb, bv, 1.f,    Vb, ROWS, D_MODEL, D_MODEL);

  attention_kernel<<<dim3(SEQ / 64, BATCH * NHEAD), 128, 0, stream>>>(Qb, Kb, Vb, msk, Ctxb);

  gemm_xwT<false, false><<<gP, blk, 0, stream>>>(Ctxb, Wob, bo, 1.f, attn, ROWS, D_MODEL, D_MODEL);
  add_norm<true ><<<ROWS, 256, 0, stream>>>(src, attn, alpha1, beta1, x1, x1b);

  gemm_xwT<true,  true ><<<dim3(D_FF / 128, ROWS / 128), blk, 0, stream>>>(x1b, W1b, b1, 1.f, Hb, ROWS, D_FF, D_MODEL);
  gemm_xwT<false, false><<<gP, blk, 0, stream>>>(Hb, W2b, b2, 1.f, ff, ROWS, D_MODEL, D_FF);
  add_norm<false><<<ROWS, 256, 0, stream>>>(x1, ff, alpha2, beta2, (float*)d_out, nullptr);
}
